// DINRanker_37812892074146
// MI455X (gfx1250) — compile-verified, matrix operand order checked by reading, not run on hardware
//
#include <hip/hip_runtime.h>
#include <hip/hip_bf16.h>
#include <math.h>
#include <stdint.h>

#define BB   4096
#define LL   200
#define LPAD 224
#define DD   64

typedef __attribute__((ext_vector_type(16))) _Float16 v16h;
typedef __attribute__((ext_vector_type(8)))  _Float16 h8;
typedef __attribute__((ext_vector_type(8)))  float    v8f;

// Load a 16x32 (MxK) f16 fragment from row-major LDS matrix (A operand), or a
// 32x16 (KxN) B operand stored transposed as [N][K] row-major.
// ISA 7.12.2 16-bit A layout: lanes 0-15 -> M=lane, K in {kh*8..kh*8+7} (VGPR0-3)
// and {16+kh*8..+7} (VGPR4-7), kh = lane>=16.
__device__ __forceinline__ v16h ld_frag(const _Float16* base, int ld, int r0, int c0, int lane) {
  int r  = r0 + (lane & 15);
  int kb = c0 + ((lane >> 4) << 3);
  const h8* p = (const h8*)(base + r * ld + kb);
  h8 lo = p[0];   // K = kb .. kb+7
  h8 hi = p[2];   // K = kb+16 .. kb+23
  return __builtin_shufflevector(lo, hi, 0,1,2,3,4,5,6,7,8,9,10,11,12,13,14,15);
}

// C/D layout: VGPR r -> M = r + 8*(lane>=16), N = lane&15. Add bias, relu, f16 store.
__device__ __forceinline__ void st_relu(_Float16* dst, int ldc, int n0,
                                        const float* bias, v8f acc, int lane) {
  int n  = lane & 15;
  int mh = (lane >> 4) << 3;
  float bv = bias[n0 + n];
  #pragma unroll
  for (int r = 0; r < 8; ++r) {
    float v = acc[r] + bv;
    dst[(mh + r) * ldc + n0 + n] = (_Float16)(v > 0.f ? v : 0.f);
  }
}

__global__ __launch_bounds__(256) void din_attention(
    const float* __restrict__ i_emb, const float* __restrict__ hist,
    const int* __restrict__ mask,
    const float* __restrict__ aW1, const float* __restrict__ ab1,
    const float* __restrict__ aW2, const float* __restrict__ ab2,
    const float* __restrict__ aW3, const float* __restrict__ ab3,
    float* __restrict__ hist_rep_out)
{
  __shared__ __align__(16) float    sHistF[LPAD * DD];    // 56 KB, async-filled
  __shared__ __align__(16) _Float16 sW1t[64 * 128];       // folded W1, B^T layout
  __shared__ __align__(16) _Float16 sW2t[32 * 64];
  __shared__ __align__(16) _Float16 sA[2][16 * 128];
  __shared__ __align__(16) _Float16 sH1[2][16 * 64];
  __shared__ __align__(16) _Float16 sH2[2][16 * 32];
  __shared__ float    st[DD];
  __shared__ __align__(16) _Float16 st16[DD];
  __shared__ float    t1b[DD];        // per-batch folded layer1 bias
  __shared__ float    sb2[32];
  __shared__ float    sW3[32];
  __shared__ float    logits[LPAD];
  __shared__ float    warr[LPAD];
  __shared__ int      smask[LPAD];
  __shared__ float    red[256];

  const int tid  = threadIdx.x;
  const int lane = tid & 31;
  const int wave = tid >> 5;
  const int b    = blockIdx.x;

  // ---------- phase 1: stage everything into LDS ----------
  // hist: 200x64 f32 via CDNA5 async global->LDS DMA path (no VGPR round trip)
  {
    uint64_t gbase  = (uint64_t)(uintptr_t)(hist + (size_t)b * LL * DD);
    uint32_t ldsbase = (uint32_t)(uintptr_t)&sHistF[0];
    for (int i = tid; i < (LL * DD) / 4; i += 256) {     // 3200 x 16B chunks
      uint32_t voff   = (uint32_t)i * 16u;
      uint32_t ldsoff = ldsbase + voff;
      asm volatile("global_load_async_to_lds_b128 %0, %1, %2"
                   :: "v"(ldsoff), "v"(voff), "s"(gbase) : "memory");
    }
  }
  if (tid < DD) {
    float v = i_emb[(size_t)b * DD + tid];
    st[tid]   = v;
    st16[tid] = (_Float16)v;
  }
  if (tid < LPAD) smask[tid] = (tid < LL) ? mask[(size_t)b * LL + tid] : 0;
  if (tid < 32) { sb2[tid] = ab2[tid]; sW3[tid] = aW3[tid]; }
  for (int i = tid; i < (LPAD - LL) * DD; i += 256) sHistF[LL * DD + i] = 0.f;

  // folded W1: k<64 -> W1a+W1d (hist), k>=64 -> W1c (hist*tgt); stored [n][k]
  for (int i = tid; i < 64 * 128; i += 256) {
    int n = i & 63, k = i >> 6;
    float v = (k < 64) ? (aW1[k * 64 + n] + aW1[(192 + k) * 64 + n])
                       : aW1[(128 + (k - 64)) * 64 + n];
    sW1t[n * 128 + k] = (_Float16)v;
  }
  for (int i = tid; i < 32 * 64; i += 256) {
    int n = i & 31, k = i >> 5;
    sW2t[n * 64 + k] = (_Float16)aW2[k * 32 + n];
  }
  asm volatile("s_wait_asynccnt 0x0" ::: "memory");
  __syncthreads();

  // ---------- phase 2: fold tgt@(W1b - W1d) + b1 into bias ----------
  if (tid < DD) {
    float acc = ab1[tid];
    for (int k = 0; k < DD; ++k)
      acc += st[k] * (aW1[(64 + k) * 64 + tid] - aW1[(192 + k) * 64 + tid]);
    t1b[tid] = acc;
  }
  __syncthreads();

  // ---------- main loop: 7 iters x 2 row-tiles of 16 ----------
  for (int it = 0; it < 7; ++it) {
    const int row0 = it * 32;
    // A tiles: [hist | hist*tgt] f16, built in 8-half vector chunks
    for (int c = tid; c < 32 * 16; c += 256) {     // 512 chunks of 8 halves
      int r   = c >> 4;            // row 0..31
      int kc  = (c & 15) << 3;     // k chunk start: 0,8,...,120
      int col = kc & 63;
      const float4* hp = (const float4*)&sHistF[(row0 + r) * DD + col];
      float4 h0 = hp[0], h1 = hp[1];
      h8 v;
      v[0] = (_Float16)h0.x; v[1] = (_Float16)h0.y;
      v[2] = (_Float16)h0.z; v[3] = (_Float16)h0.w;
      v[4] = (_Float16)h1.x; v[5] = (_Float16)h1.y;
      v[6] = (_Float16)h1.z; v[7] = (_Float16)h1.w;
      if (kc >= 64) v *= *(const h8*)&st16[col];   // hist (.) tgt half
      *(h8*)&sA[r >> 4][(r & 15) * 128 + kc] = v;
    }
    __syncthreads();
    // layer1: [16x128] x [128x64], 8 waves = 2 tiles x 4 n-tiles
    {
      int t = wave >> 2, nt = wave & 3;
      v8f acc = {};
      #pragma unroll
      for (int kk = 0; kk < 4; ++kk) {
        v16h a  = ld_frag(sA[t], 128, 0, kk * 32, lane);
        v16h bm = ld_frag(sW1t, 128, nt * 16, kk * 32, lane);
        acc = __builtin_amdgcn_wmma_f32_16x16x32_f16(false, a, false, bm, (short)0, acc, false, false);
      }
      st_relu(sH1[t], 64, nt * 16, t1b, acc, lane);
    }
    __syncthreads();
    // layer2: [16x64] x [64x32]
    if (wave < 4) {
      int t = wave >> 1, nt = wave & 1;
      v8f acc = {};
      #pragma unroll
      for (int kk = 0; kk < 2; ++kk) {
        v16h a  = ld_frag(sH1[t], 64, 0, kk * 32, lane);
        v16h bm = ld_frag(sW2t, 64, nt * 16, kk * 32, lane);
        acc = __builtin_amdgcn_wmma_f32_16x16x32_f16(false, a, false, bm, (short)0, acc, false, false);
      }
      st_relu(sH2[t], 32, nt * 16, sb2, acc, lane);
    }
    __syncthreads();
    // layer3: 32 logits via VALU dot
    if (tid < 32) {
      float acc = ab3[0];
      const _Float16* h2 = sH2[tid >> 4] + (tid & 15) * 32;
      #pragma unroll
      for (int k = 0; k < 32; ++k) acc += (float)h2[k] * sW3[k];
      logits[row0 + tid] = acc;
    }
    __syncthreads();
  }

  // ---------- masked softmax over L ----------
  bool on = (tid < LPAD) && (smask[tid] != 0);
  float lv = on ? logits[tid] : -__builtin_huge_valf();
  red[tid] = lv;
  __syncthreads();
  for (int off = 128; off > 0; off >>= 1) {
    if (tid < off) red[tid] = fmaxf(red[tid], red[tid + off]);
    __syncthreads();
  }
  float mx = red[0];
  __syncthreads();
  float e = on ? __expf(logits[tid] - mx) : 0.f;
  red[tid] = e;
  __syncthreads();
  for (int off = 128; off > 0; off >>= 1) {
    if (tid < off) red[tid] += red[tid + off];
    __syncthreads();
  }
  float s = red[0];
  __syncthreads();
  if (tid < LPAD) warr[tid] = (s > 0.f) ? (e / s) : 0.f;
  __syncthreads();

  // ---------- hist_rep = w @ hist (f32 LDS copy, no 2nd HBM read) ----------
  {
    int d = tid & 63, part = tid >> 6;
    float acc = 0.f;
    for (int l = part; l < LPAD; l += 4) acc += warr[l] * sHistF[l * DD + d];
    red[tid] = acc;
    __syncthreads();
    if (part == 0)
      hist_rep_out[(size_t)b * DD + d] = red[d] + red[64 + d] + red[128 + d] + red[192 + d];
  }
}

__global__ __launch_bounds__(256) void din_dnn(
    const float* __restrict__ u_emb, const float* __restrict__ i_emb,
    const float* __restrict__ hist_rep,
    const float* __restrict__ dW1, const float* __restrict__ db1,
    const float* __restrict__ dW2, const float* __restrict__ db2,
    const float* __restrict__ oW,  const float* __restrict__ ob,
    float* __restrict__ out)
{
  __shared__ __align__(16) _Float16 sX[16 * 192];
  __shared__ __align__(16) _Float16 sW1t[128 * 192];   // [n][k]
  __shared__ __align__(16) _Float16 sW2t[64 * 128];
  __shared__ __align__(16) _Float16 sH1[16 * 128];
  __shared__ __align__(16) _Float16 sH2[16 * 64];
  __shared__ float sb1[128];
  __shared__ float sb2d[64];
  __shared__ float soW[64];

  const int tid  = threadIdx.x;
  const int lane = tid & 31;
  const int wave = tid >> 5;
  const int b0   = blockIdx.x * 16;

  for (int i = tid; i < 16 * 192; i += 256) {
    int r = i / 192, c = i % 192;
    size_t row = (size_t)(b0 + r) * 64;
    float v = (c < 64)  ? u_emb[row + c]
            : (c < 128) ? i_emb[row + (c - 64)]
                        : hist_rep[row + (c - 128)];
    sX[i] = (_Float16)v;
  }
  for (int i = tid; i < 128 * 192; i += 256) {
    int n = i & 127, k = i >> 7;
    sW1t[n * 192 + k] = (_Float16)dW1[k * 128 + n];
  }
  for (int i = tid; i < 64 * 128; i += 256) {
    int n = i & 63, k = i >> 6;
    sW2t[n * 128 + k] = (_Float16)dW2[k * 64 + n];
  }
  if (tid < 128) sb1[tid] = db1[tid];
  if (tid < 64)  { sb2d[tid] = db2[tid]; soW[tid] = oW[tid]; }
  __syncthreads();

  {  // layer1: [16x192] x [192x128]
    int nt = wave;
    v8f acc = {};
    #pragma unroll
    for (int kk = 0; kk < 6; ++kk) {
      v16h a  = ld_frag(sX, 192, 0, kk * 32, lane);
      v16h bm = ld_frag(sW1t, 192, nt * 16, kk * 32, lane);
      acc = __builtin_amdgcn_wmma_f32_16x16x32_f16(false, a, false, bm, (short)0, acc, false, false);
    }
    st_relu(sH1, 128, nt * 16, sb1, acc, lane);
  }
  __syncthreads();
  if (wave < 4) {  // layer2: [16x128] x [128x64]
    int nt = wave;
    v8f acc = {};
    #pragma unroll
    for (int kk = 0; kk < 4; ++kk) {
      v16h a  = ld_frag(sH1, 128, 0, kk * 32, lane);
      v16h bm = ld_frag(sW2t, 128, nt * 16, kk * 32, lane);
      acc = __builtin_amdgcn_wmma_f32_16x16x32_f16(false, a, false, bm, (short)0, acc, false, false);
    }
    st_relu(sH2, 64, nt * 16, sb2d, acc, lane);
  }
  __syncthreads();
  if (tid < 16) {  // output + sigmoid
    float acc = ob[0];
    #pragma unroll
    for (int k = 0; k < 64; ++k) acc += (float)sH2[tid * 64 + k] * soW[k];
    out[b0 + tid] = 1.f / (1.f + __expf(-acc));
  }
}

extern "C" void kernel_launch(void* const* d_in, const int* in_sizes, int n_in,
                              void* d_out, int out_size, void* d_ws, size_t ws_size,
                              hipStream_t stream) {
  (void)in_sizes; (void)n_in; (void)out_size; (void)ws_size;
  const float* u    = (const float*)d_in[0];
  const float* ie   = (const float*)d_in[1];
  const float* hist = (const float*)d_in[2];
  const int*   mask = (const int*)d_in[3];
  const float* aW1 = (const float*)d_in[4],  *ab1 = (const float*)d_in[5];
  const float* aW2 = (const float*)d_in[6],  *ab2 = (const float*)d_in[7];
  const float* aW3 = (const float*)d_in[8],  *ab3 = (const float*)d_in[9];
  const float* dW1 = (const float*)d_in[10], *db1 = (const float*)d_in[11];
  const float* dW2 = (const float*)d_in[12], *db2 = (const float*)d_in[13];
  const float* oW  = (const float*)d_in[14], *ob  = (const float*)d_in[15];
  float* hist_rep = (float*)d_ws;          // BB * DD floats (1 MB)
  float* out      = (float*)d_out;

  din_attention<<<BB, 256, 0, stream>>>(ie, hist, mask,
                                        aW1, ab1, aW2, ab2, aW3, ab3, hist_rep);
  din_dnn<<<BB / 16, 256, 0, stream>>>(u, ie, hist_rep,
                                       dW1, db1, dW2, db2, oW, ob, out);
}
